// RecGraphTransformerLayer_60790967108360
// MI455X (gfx1250) — matrix operand order repaired and checked by reading, not compiled, not run on hardware
//
#include <hip/hip_runtime.h>
#include <stdint.h>

// ---------------------------------------------------------------------------
// Types for CDNA5 WMMA (wave32): v_wmma_f32_16x16x32_bf16
// ---------------------------------------------------------------------------
typedef __attribute__((ext_vector_type(16))) __bf16 bfx16;
typedef __attribute__((ext_vector_type(8)))  __bf16 bfx8;
typedef __attribute__((ext_vector_type(8)))  float  fx8;

#define NNODES 50000
#define DMODEL 128

// ---------------------------------------------------------------------------
// f32 -> bf16 conversion (weights once per launch, activations as needed)
// ---------------------------------------------------------------------------
__global__ void k_f32_to_bf16(const float* __restrict__ x, __bf16* __restrict__ y, int n) {
    int i = blockIdx.x * blockDim.x + threadIdx.x;
    if (i < n) y[i] = (__bf16)x[i];
}

// ---------------------------------------------------------------------------
// GEMM: C[N,M] = act(A[N,K](bf16) @ W[M,K](bf16)^T + bias)
// Block: 256 threads = 8 waves. Block tile 128 rows x 64 cols.
// Wave w -> rows 16w..16w+15, 4 col-tiles of 16. K-loop step 32.
// A fragment per CDNA5 ISA 16-bit A 16x32 layout; B fragment (same k split as A
// per lane) read from LDS-staged W slice (row-major [64][K], padded).
// act: 0=none 1=relu 2=sigmoid.  Cf (f32) and Cbf (bf16) optional outputs.
// ---------------------------------------------------------------------------
__global__ __launch_bounds__(256)
void k_gemm_bf16_wmma(const __bf16* __restrict__ A, const __bf16* __restrict__ W,
                      const float* __restrict__ bias, float* __restrict__ Cf,
                      __bf16* __restrict__ Cbf, int N, int K, int M, int act)
{
    __shared__ __bf16 sW[64 * 272];                 // 64 rows x (Kmax=256 + 16 pad)
    const int tid  = threadIdx.x;
    const int wave = tid >> 5;
    const int lane = tid & 31;
    const int KP   = K + 16;                        // padded row stride (16B-aligned, 8-bank skew)
    const int colBase = blockIdx.y * 64;

    // Stage W slice [64, K] into LDS
    for (int i = tid; i < 64 * K; i += 256) {
        int c = i / K, k = i - c * K;
        sW[c * KP + k] = W[(size_t)(colBase + c) * K + k];
    }
    __syncthreads();

    const int rowBase = blockIdx.x * 128 + wave * 16;
    const int lrow = lane & 15;
    const int hsel = lane >> 4;                     // 0: lanes 0-15, 1: lanes 16-31
    int arow = rowBase + lrow; if (arow >= N) arow = N - 1;   // clamp (stores guarded)
    const int kA0 = hsel * 8;                       // A elems 0..7 -> K+kA0.., 8..15 -> K+kA0+16..
    const int kB0 = hsel * 8;                       // B elems 0..7 -> K+kB0.., 8..15 -> K+kB0+16..

    fx8 acc[4]; 
    #pragma unroll
    for (int t = 0; t < 4; ++t) acc[t] = (fx8){};

    const __bf16* __restrict__ arowp = A + (size_t)arow * K;

    for (int ko = 0; ko < K; ko += 32) {
        bfx8 alo = *(const bfx8*)(arowp + ko + kA0);
        bfx8 ahi = *(const bfx8*)(arowp + ko + kA0 + 16);
        bfx16 a;
        #pragma unroll
        for (int i = 0; i < 8; ++i) { a[i] = alo[i]; a[8 + i] = ahi[i]; }
        #pragma unroll
        for (int t = 0; t < 4; ++t) {
            const __bf16* bp = &sW[(t * 16 + lrow) * KP + ko + kB0];
            bfx8 blo = *(const bfx8*)(bp);
            bfx8 bhi = *(const bfx8*)(bp + 16);
            bfx16 b;
            #pragma unroll
            for (int i = 0; i < 8; ++i) { b[i] = blo[i]; b[8 + i] = bhi[i]; }
            acc[t] = __builtin_amdgcn_wmma_f32_16x16x32_bf16(
                false, a, false, b, (short)0, acc[t], false, false);
        }
    }

    // Epilogue: bias + activation; C layout: VGPR i -> row rowBase+i+8*hsel, col = tile+lrow
    #pragma unroll
    for (int t = 0; t < 4; ++t) {
        const int col = colBase + t * 16 + lrow;
        const float bv = bias ? bias[col] : 0.f;
        #pragma unroll
        for (int i = 0; i < 8; ++i) {
            int row = rowBase + i + hsel * 8;
            if (row < N) {
                float v = acc[t][i] + bv;
                if (act == 1) v = fmaxf(v, 0.f);
                else if (act == 2) v = 1.f / (1.f + __expf(-v));
                if (Cf)  Cf [(size_t)row * M + col] = v;
                if (Cbf) Cbf[(size_t)row * M + col] = (__bf16)v;
            }
        }
    }
}

// ---------------------------------------------------------------------------
// Edge attention: one thread per (edge, head). score = exp(clip(<K[s],Q[d]>/4))
// scatter: z[d,h] += score ; wV[d, h*16..] += score * V[s, h*16..]
// ---------------------------------------------------------------------------
__global__ __launch_bounds__(256)
void k_edge_attn(const float* __restrict__ Q, const float* __restrict__ K,
                 const float* __restrict__ V, const int* __restrict__ src,
                 const int* __restrict__ dst, float* __restrict__ wV,
                 float* __restrict__ z, int E)
{
    long long idx = (long long)blockIdx.x * blockDim.x + threadIdx.x;
    if (idx >= (long long)E * 8) return;
    int e = (int)(idx >> 3);
    int h = (int)(idx & 7);
    int s = src[e], d = dst[e];
    const float4* kp = (const float4*)(K + (size_t)s * 128 + h * 16);
    const float4* qp = (const float4*)(Q + (size_t)d * 128 + h * 16);
    const float4* vp = (const float4*)(V + (size_t)s * 128 + h * 16);
    __builtin_prefetch(vp, 0, 0);                   // global_prefetch_b8: hide V gather
    float dot = 0.f;
    #pragma unroll
    for (int j = 0; j < 4; ++j) {
        float4 kv = kp[j], qv = qp[j];
        dot += kv.x * qv.x + kv.y * qv.y + kv.z * qv.z + kv.w * qv.w;
    }
    float sc = __expf(fminf(fmaxf(dot * 0.25f, -5.f), 5.f));   // 1/sqrt(16)=0.25
    atomicAdd(&z[(size_t)d * 8 + h], sc);
    float* w = wV + (size_t)d * 128 + h * 16;
    #pragma unroll
    for (int j = 0; j < 4; ++j) {
        float4 vv = vp[j];
        atomicAdd(&w[4 * j + 0], sc * vv.x);
        atomicAdd(&w[4 * j + 1], sc * vv.y);
        atomicAdd(&w[4 * j + 2], sc * vv.z);
        atomicAdd(&w[4 * j + 3], sc * vv.w);
    }
}

// ---------------------------------------------------------------------------
// Normalize (wV/z) for both attentions and write concatenated bf16 [N,256]
// ---------------------------------------------------------------------------
__global__ void k_concat_norm(const float* __restrict__ wVa, const float* __restrict__ za,
                              const float* __restrict__ wVb, const float* __restrict__ zb,
                              __bf16* __restrict__ cat, int N)
{
    int idx = blockIdx.x * blockDim.x + threadIdx.x;
    if (idx >= N * 128) return;
    int n = idx >> 7, c = idx & 127, h = c >> 4;
    cat[(size_t)n * 256 + c]       = (__bf16)(wVa[idx] / za[n * 8 + h]);
    cat[(size_t)n * 256 + 128 + c] = (__bf16)(wVb[idx] / zb[n * 8 + h]);
}

// ---------------------------------------------------------------------------
// Residual / gated combine (pre-BN input)
// ---------------------------------------------------------------------------
__global__ void k_combine_add(const float* __restrict__ a, const float* __restrict__ b,
                              float* __restrict__ x, int n)
{
    int i = blockIdx.x * blockDim.x + threadIdx.x;
    if (i < n) x[i] = a[i] + b[i];
}
__global__ void k_combine_gate(const float* __restrict__ a, const float* __restrict__ b,
                               const float* __restrict__ g, float* __restrict__ x, int n)
{
    int i = blockIdx.x * blockDim.x + threadIdx.x;
    if (i < n) { float gg = g[i]; x[i] = (1.f - gg) * a[i] + gg * b[i]; }
}

// ---------------------------------------------------------------------------
// BatchNorm (training mode, biased stats): two-pass
// ---------------------------------------------------------------------------
__global__ __launch_bounds__(128)
void k_bn_stats(const float* __restrict__ x, float* __restrict__ sums, int N)
{
    int c = threadIdx.x;                 // 128 channels
    float s = 0.f, s2 = 0.f;
    for (int n = blockIdx.x; n < N; n += gridDim.x) {
        float v = x[(size_t)n * 128 + c];
        s += v; s2 += v * v;
    }
    atomicAdd(&sums[c], s);
    atomicAdd(&sums[128 + c], s2);
}
__global__ void k_bn_apply(const float* __restrict__ x, const float* __restrict__ sums,
                           const float* __restrict__ gamma, const float* __restrict__ beta,
                           float* __restrict__ yf, __bf16* __restrict__ ybf, int N)
{
    int idx = blockIdx.x * blockDim.x + threadIdx.x;
    if (idx >= N * 128) return;
    int c = idx & 127;
    const float invN = 1.f / (float)N;
    float mean = sums[c] * invN;
    float var  = sums[128 + c] * invN - mean * mean;
    float v = (x[idx] - mean) * rsqrtf(var + 1e-5f) * gamma[c] + beta[c];
    if (yf)  yf[idx]  = v;
    if (ybf) ybf[idx] = (__bf16)v;
}

// ---------------------------------------------------------------------------
// Host orchestration
// ---------------------------------------------------------------------------
extern "C" void kernel_launch(void* const* d_in, const int* in_sizes, int n_in,
                              void* d_out, int out_size, void* d_ws, size_t ws_size,
                              hipStream_t stream)
{
    const int N = NNODES, D = DMODEL;
    const int E = in_sizes[2];
    const int ND = N * D;

    const float* h_in  = (const float*)d_in[0];
    const float* sv_in = (const float*)d_in[1];
    const int*   src   = (const int*)d_in[2];
    const int*   dst   = (const int*)d_in[3];
    const float* bo    = (const float*)d_in[14];
    const float* bo_h  = (const float*)d_in[16];
    const float* b1    = (const float*)d_in[18];
    const float* b2    = (const float*)d_in[20];
    const float* b1h   = (const float*)d_in[22];
    const float* b2h   = (const float*)d_in[24];
    const float* bg1   = (const float*)d_in[26];
    const float* bg2   = (const float*)d_in[28];
    const float* gbn1  = (const float*)d_in[29]; const float* bbn1  = (const float*)d_in[30];
    const float* gbn2  = (const float*)d_in[31]; const float* bbn2  = (const float*)d_in[32];
    const float* gbn1h = (const float*)d_in[33]; const float* bbn1h = (const float*)d_in[34];
    const float* gbn2h = (const float*)d_in[35]; const float* bbn2h = (const float*)d_in[36];

    // ---- workspace carve-out (256B aligned) ----
    char* base = (char*)d_ws;
    size_t off = 0;
    auto alloc = [&](size_t bytes) -> void* {
        void* p = base + off;
        off = (off + bytes + 255) & ~(size_t)255;
        return p;
    };

    // bf16 weights: 0..8 = Wq_sv..Wv_sh (128x128); 9=Wo 10=Wo_h (128x256);
    // 11=W1 (256x128); 12=W2 (128x256); 13=W1h (256x128); 14=W2h (128x256);
    // 15=Wg1 16=Wg2 (128x128)
    const int widx[17] = {4,5,6,7,8,9,10,11,12, 13,15, 17,19,21,23, 25,27};
    const int wn[17]   = {16384,16384,16384,16384,16384,16384,16384,16384,16384,
                          32768,32768, 32768,32768,32768,32768, 16384,16384};
    __bf16* WB[17];
    for (int i = 0; i < 17; ++i) WB[i] = (__bf16*)alloc((size_t)wn[i] * 2);

    __bf16* hbf   = (__bf16*)alloc((size_t)ND * 2);
    __bf16* svbf  = (__bf16*)alloc((size_t)ND * 2);
    float*  Qf    = (float*)alloc((size_t)ND * 4);
    float*  Kf    = (float*)alloc((size_t)ND * 4);
    float*  Vf    = (float*)alloc((size_t)ND * 4);
    float*  wVa   = (float*)alloc((size_t)ND * 4);
    float*  wVb   = (float*)alloc((size_t)ND * 4);
    float*  za    = (float*)alloc((size_t)N * 8 * 4);
    float*  zb    = (float*)alloc((size_t)N * 8 * 4);
    __bf16* catbf = (__bf16*)alloc((size_t)N * 256 * 2);
    float*  bufO  = (float*)alloc((size_t)ND * 4);
    float*  bufX  = (float*)alloc((size_t)ND * 4);
    float*  bufY  = (float*)alloc((size_t)ND * 4);
    __bf16* bufYbf= (__bf16*)alloc((size_t)ND * 2);
    __bf16* tbf   = (__bf16*)alloc((size_t)N * 256 * 2);
    float*  bufF  = (float*)alloc((size_t)ND * 4);
    float*  bufG  = (float*)alloc((size_t)ND * 4);
    float*  bnsum = (float*)alloc(256 * 4);

    float* out_h  = (float*)d_out;
    float* out_sv = (float*)d_out + (size_t)ND;

    // ---- helpers ----
    auto cvt = [&](const float* s, __bf16* d, int n) {
        k_f32_to_bf16<<<(n + 255) / 256, 256, 0, stream>>>(s, d, n);
    };
    auto gemm = [&](const __bf16* A, const __bf16* W, const float* bias,
                    float* Cf, __bf16* Cbf, int K, int M, int act) {
        dim3 g((N + 127) / 128, M / 64);
        k_gemm_bf16_wmma<<<g, 256, 0, stream>>>(A, W, bias, Cf, Cbf, N, K, M, act);
    };
    auto edge = [&](float* wV, float* z) {
        hipMemsetAsync(wV, 0, (size_t)ND * 4, stream);
        hipMemsetAsync(z,  0, (size_t)N * 8 * 4, stream);
        long long tot = (long long)E * 8;
        k_edge_attn<<<(unsigned)((tot + 255) / 256), 256, 0, stream>>>(Qf, Kf, Vf, src, dst, wV, z, E);
    };
    auto bnorm = [&](const float* x, const float* gamma, const float* beta,
                     float* yf, __bf16* ybf) {
        hipMemsetAsync(bnsum, 0, 256 * 4, stream);
        k_bn_stats<<<512, 128, 0, stream>>>(x, bnsum, N);
        k_bn_apply<<<(ND + 255) / 256, 256, 0, stream>>>(x, bnsum, gamma, beta, yf, ybf, N);
    };

    // ---- precision conversion ----
    for (int i = 0; i < 17; ++i) cvt((const float*)d_in[widx[i]], WB[i], wn[i]);
    cvt(h_in, hbf, ND);
    cvt(sv_in, svbf, ND);

    // ================= h branch =================
    // a_self = attn(h,h,h ; Wq_sv,Wk_sv,Wv_sv)
    gemm(hbf, WB[0], nullptr, Qf, nullptr, 128, 128, 0);
    gemm(hbf, WB[1], nullptr, Kf, nullptr, 128, 128, 0);
    gemm(hbf, WB[2], nullptr, Vf, nullptr, 128, 128, 0);
    edge(wVa, za);
    // a_cross = attn(h,sv,sv ; Wq_cv,Wk_cv,Wv_cv)
    gemm(hbf,  WB[3], nullptr, Qf, nullptr, 128, 128, 0);
    gemm(svbf, WB[4], nullptr, Kf, nullptr, 128, 128, 0);
    gemm(svbf, WB[5], nullptr, Vf, nullptr, 128, 128, 0);
    edge(wVb, zb);
    // hO = concat @ Wo^T + bo
    k_concat_norm<<<(ND + 255) / 256, 256, 0, stream>>>(wVa, za, wVb, zb, catbf, N);
    gemm(catbf, WB[9], bo, bufO, nullptr, 256, 128, 0);
    // h1 = BN(h + hO)
    k_combine_add<<<(ND + 255) / 256, 256, 0, stream>>>(h_in, bufO, bufX, ND);
    bnorm(bufX, gbn1, bbn1, bufY, bufYbf);
    // FFN + BN
    gemm(bufYbf, WB[11], b1, nullptr, tbf, 128, 256, 1);   // relu
    gemm(tbf,    WB[12], b2, bufF, nullptr, 256, 128, 0);
    k_combine_add<<<(ND + 255) / 256, 256, 0, stream>>>(bufY, bufF, bufX, ND);
    bnorm(bufX, gbn2, bbn2, out_h, nullptr);

    // ================= sv branch =================
    // a_self_h = attn(sv,sv,sv ; Wq_sh,Wk_sh,Wv_sh)
    gemm(svbf, WB[6], nullptr, Qf, nullptr, 128, 128, 0);
    gemm(svbf, WB[7], nullptr, Kf, nullptr, 128, 128, 0);
    gemm(svbf, WB[8], nullptr, Vf, nullptr, 128, 128, 0);
    edge(wVa, za);
    // a_cross_h = attn(sv,h,h ; Wq_cv,Wk_cv,Wv_cv)  (weights reused, faithful to ref)
    gemm(svbf, WB[3], nullptr, Qf, nullptr, 128, 128, 0);
    gemm(hbf,  WB[4], nullptr, Kf, nullptr, 128, 128, 0);
    gemm(hbf,  WB[5], nullptr, Vf, nullptr, 128, 128, 0);
    edge(wVb, zb);
    // svO = concat @ Wo_h^T + bo_h
    k_concat_norm<<<(ND + 255) / 256, 256, 0, stream>>>(wVa, za, wVb, zb, catbf, N);
    gemm(catbf, WB[10], bo_h, bufO, nullptr, 256, 128, 0);
    // g1 gate from ORIGINAL h
    gemm(hbf, WB[15], bg1, bufG, nullptr, 128, 128, 2);    // sigmoid
    k_combine_gate<<<(ND + 255) / 256, 256, 0, stream>>>(sv_in, bufO, bufG, bufX, ND);
    bnorm(bufX, gbn1h, bbn1h, bufY, bufYbf);               // sv1
    // FFN_h
    gemm(bufYbf, WB[13], b1h, nullptr, tbf, 128, 256, 1);  // relu
    gemm(tbf,    WB[14], b2h, bufF, nullptr, 256, 128, 0);
    // g2 gate from ORIGINAL h
    gemm(hbf, WB[16], bg2, bufG, nullptr, 128, 128, 2);    // sigmoid
    k_combine_gate<<<(ND + 255) / 256, 256, 0, stream>>>(bufY, bufF, bufG, bufX, ND);
    bnorm(bufX, gbn2h, bbn2h, out_sv, nullptr);
}
